// MomentumVectorQuantize_19963007992474
// MI455X (gfx1250) — matrix operand order repaired and checked by reading, not compiled, not run on hardware
//
#include <hip/hip_runtime.h>

typedef __attribute__((ext_vector_type(16))) _Float16 v16h;
typedef __attribute__((ext_vector_type(8)))  float    v8f;

#define D_DIM   256
#define K_CODES 1024
#define N_TOK   65536
#define HW      4096
#define CH      64        // codes staged per LDS chunk
#define RS      264       // LDS row stride in halves (256 + 8 pad, 16B aligned)

// workspace layout (byte offsets)
#define WS_ENORM 0
#define WS_EMBH  4096
#define WS_IND   (4096 + 524288)
#define WS_PART  (WS_IND + 262144)

// ---------------------------------------------------------------------------
// Kernel 0: codebook -> f16 transposed [code][d], and ||e||^2 per code
// (norms from the f16-rounded values so the quadratic expansion used for the
//  argmin is self-consistent).
// ---------------------------------------------------------------------------
__global__ __launch_bounds__(256) void k_prep(const float* __restrict__ embed,
                                              _Float16* __restrict__ embh,
                                              float* __restrict__ enorm) {
  const int k = blockIdx.x;   // code
  const int d = threadIdx.x;  // dim
  const float v = embed[(size_t)d * K_CODES + k];
  const _Float16 h = (_Float16)v;
  embh[(size_t)k * D_DIM + d] = h;
  const float hv = (float)h;
  __shared__ float red[256];
  red[d] = hv * hv;
  __syncthreads();
  for (int s = 128; s > 0; s >>= 1) {
    if (d < s) red[d] += red[d + s];
    __syncthreads();
  }
  if (d == 0) enorm[k] = red[0];
}

// ---------------------------------------------------------------------------
// Kernel 1: per-token argmin over 1024 codes via f16 WMMA (f32 accumulate).
// 8 waves/block, 32 tokens/wave (two 16-row A tiles -> each LDS B fragment
// feeds 2 WMMAs). Codebook staged through LDS in 64-code chunks with a
// register prefetch pipeline so chunk c+1 global loads overlap chunk c math.
// dist = ||e||^2 - 2*(x . e)  (||x||^2 dropped: constant per row).
// ---------------------------------------------------------------------------
__global__ __launch_bounds__(256) void k_argmin(const float* __restrict__ x,
                                                const _Float16* __restrict__ embh,
                                                const float* __restrict__ enorm,
                                                int* __restrict__ ind) {
  __shared__ _Float16 lds[CH * RS];
  const int tid  = threadIdx.x;
  const int wave = tid >> 5;
  const int lane = tid & 31;
  const int row  = lane & 15;      // A-matrix M index / B-matrix N index
  const int hi   = lane >> 4;      // 0: lanes 0-15, 1: lanes 16-31

  const int n0  = blockIdx.x * 256;   // block token base
  const int tw0 = n0 + wave * 32;     // wave token base (two 16-token tiles)

  // ---- A fragments: 2 x (16 tokens x 256 d) in ISA 16-bit A 16x32 layout --
  // lanes 0-15: halves 0..7 = K kb+0..7, halves 8..15 = K kb+16..23
  // lanes 16-31: same with kbase = 8
  v16h a[2][8];
  const int kbase = hi ? 8 : 0;
#pragma unroll
  for (int mt = 0; mt < 2; ++mt) {
    const int tok = tw0 + mt * 16 + row;
    const int b   = tok >> 12;
    const int hw  = tok & (HW - 1);
    const float* xtok = x + (size_t)b * D_DIM * HW + hw;  // NCHW: stride HW per d
#pragma unroll
    for (int f = 0; f < 8; ++f) {
      const int kb = f * 32 + kbase;
#pragma unroll
      for (int j = 0; j < 8; ++j) {
        a[mt][f][j]     = (_Float16)xtok[(size_t)(kb + j) * HW];
        a[mt][f][j + 8] = (_Float16)xtok[(size_t)(kb + 16 + j) * HW];
      }
    }
  }

  float best[2][8];
  int   bidx[2][8];
#pragma unroll
  for (int mt = 0; mt < 2; ++mt)
#pragma unroll
    for (int j = 0; j < 8; ++j) { best[mt][j] = 3.0e38f; bidx[mt][j] = 0; }

  // staging pipeline: this thread copies 8 uint4 (128B) per chunk
  const int    cc  = tid >> 2;
  const int    seg = tid & 3;
  const uint4* embu  = (const uint4*)embh;
  const size_t gbase = (size_t)cc * 32 + seg * 8;   // uint4 index within chunk
  uint4*       ldst  = (uint4*)(lds + cc * RS + seg * 64);

  uint4 pre[8];
#pragma unroll
  for (int i = 0; i < 8; ++i) pre[i] = embu[gbase + i];   // chunk 0

  for (int c0 = 0; c0 < K_CODES; c0 += CH) {
    __syncthreads();                 // previous chunk's compute done
#pragma unroll
    for (int i = 0; i < 8; ++i) ldst[i] = pre[i];
    __syncthreads();

    if (c0 + CH < K_CODES) {         // prefetch next chunk behind the math
      const size_t g = gbase + (size_t)(c0 + CH) * 32;
#pragma unroll
      for (int i = 0; i < 8; ++i) pre[i] = embu[g + i];
    }

#pragma unroll
    for (int nt = 0; nt < CH / 16; ++nt) {
      const int code = c0 + nt * 16 + row;
      v8f acc0 = {};
      v8f acc1 = {};
#pragma unroll
      for (int f = 0; f < 8; ++f) {
        // B 32x16 layout (matches A): lanes 0-15 hold K 0..7,16..23 (col N=lane),
        // lanes 16-31 hold K 8..15,24..31 -> two 8-half chunks of our LDS row.
        union { uint4 u[2]; v16h h; } bf;
        const _Float16* bp = lds + (nt * 16 + row) * RS + f * 32 + hi * 8;
        bf.u[0] = *(const uint4*)(bp);
        bf.u[1] = *(const uint4*)(bp + 16);
        acc0 = __builtin_amdgcn_wmma_f32_16x16x32_f16(false, a[0][f], false, bf.h,
                                                      (short)0, acc0, false, false);
        acc1 = __builtin_amdgcn_wmma_f32_16x16x32_f16(false, a[1][f], false, bf.h,
                                                      (short)0, acc1, false, false);
      }
      const float en = enorm[code];
#pragma unroll
      for (int j = 0; j < 8; ++j) {  // accN[j] = S[tile row j + 8*hi][code]
        const float d0 = en - 2.0f * acc0[j];
        if (d0 < best[0][j]) { best[0][j] = d0; bidx[0][j] = code; }
        const float d1 = en - 2.0f * acc1[j];
        if (d1 < best[1][j]) { best[1][j] = d1; bidx[1][j] = code; }
      }
    }
  }

  // reduce across the 16 lanes that share the same token rows
  // (first-min-index tie-break to match jnp.argmin)
#pragma unroll
  for (int m = 1; m <= 8; m <<= 1) {
#pragma unroll
    for (int mt = 0; mt < 2; ++mt)
#pragma unroll
      for (int j = 0; j < 8; ++j) {
        const float od = __shfl_xor(best[mt][j], m, 32);
        const int   oc = __shfl_xor(bidx[mt][j], m, 32);
        if (od < best[mt][j] || (od == best[mt][j] && oc < bidx[mt][j])) {
          best[mt][j] = od; bidx[mt][j] = oc;
        }
      }
  }
  if (row == 0) {
#pragma unroll
    for (int mt = 0; mt < 2; ++mt)
#pragma unroll
      for (int j = 0; j < 8; ++j)
        ind[tw0 + mt * 16 + hi * 8 + j] = bidx[mt][j];
  }
}

// ---------------------------------------------------------------------------
// Kernel 2: gather fp32 codebook rows into NCHW output (exact embed values),
// accumulate per-block squared-error partials for the loss.
// ---------------------------------------------------------------------------
__global__ __launch_bounds__(256) void k_out(const float* __restrict__ x,
                                             const float* __restrict__ embed,
                                             const int* __restrict__ ind,
                                             float* __restrict__ out,
                                             float* __restrict__ partial) {
  const int tid    = threadIdx.x;
  const int b      = blockIdx.x >> 6;
  const int hw0    = (blockIdx.x & 63) * 64;
  const int tokoff = tid & 63;
  const int idx    = ind[b * HW + hw0 + tokoff];
  float acc = 0.0f;
  for (int d = tid >> 6; d < D_DIM; d += 4) {
    const size_t off = (size_t)(b * D_DIM + d) * HW + hw0 + tokoff;
    const float q  = embed[(size_t)d * K_CODES + idx];  // 1MB -> L2 resident
    const float xv = x[off];
    const float df = q - xv;
    acc += df * df;
    out[off] = q;                                       // coalesced NCHW store
  }
  __shared__ float red[256];
  red[tid] = acc;
  __syncthreads();
  for (int s = 128; s > 0; s >>= 1) {
    if (tid < s) red[tid] += red[tid + s];
    __syncthreads();
  }
  if (tid == 0) partial[blockIdx.x] = red[0];
}

// ---------------------------------------------------------------------------
// Kernel 3: deterministic final reduction of 1024 partials -> loss scalar.
// ---------------------------------------------------------------------------
__global__ __launch_bounds__(256) void k_loss(const float* __restrict__ partial,
                                              float* __restrict__ loss_out) {
  __shared__ float red[256];
  const int tid = threadIdx.x;
  red[tid] = partial[tid] + partial[tid + 256] + partial[tid + 512] + partial[tid + 768];
  __syncthreads();
  for (int s = 128; s > 0; s >>= 1) {
    if (tid < s) red[tid] += red[tid + s];
    __syncthreads();
  }
  if (tid == 0)
    loss_out[0] = red[0] * (0.25f / ((float)N_TOK * (float)D_DIM));
}

extern "C" void kernel_launch(void* const* d_in, const int* in_sizes, int n_in,
                              void* d_out, int out_size, void* d_ws, size_t ws_size,
                              hipStream_t stream) {
  const float* x     = (const float*)d_in[0];   // (16,256,64,64) fp32 NCHW
  const float* embed = (const float*)d_in[1];   // (256,1024) fp32
  float* out = (float*)d_out;                   // [0]=loss, [1..]=NCHW output

  char* ws = (char*)d_ws;
  float*    enorm = (float*)(ws + WS_ENORM);
  _Float16* embh  = (_Float16*)(ws + WS_EMBH);
  int*      ind   = (int*)(ws + WS_IND);
  float*    part  = (float*)(ws + WS_PART);

  k_prep  <<<K_CODES,      256, 0, stream>>>(embed, embh, enorm);
  k_argmin<<<N_TOK / 256,  256, 0, stream>>>(x, embh, enorm, ind);
  k_out   <<<1024,         256, 0, stream>>>(x, embed, ind, out + 1, part);
  k_loss  <<<1,            256, 0, stream>>>(part, out);
}
